// SubGraph_43155831390245
// MI455X (gfx1250) — matrix operand order repaired and checked by reading, not compiled, not run on hardware
//
#include <hip/hip_runtime.h>

// ---------------------------------------------------------------------------
// Fused VectorNet subgraph encoder for MI455X (gfx1250, wave32, WMMA).
// One wave == one 32-node cluster. All GEMMs via v_wmma_f32_16x16x32_bf16.
// Weights transposed to bf16 in LDS once per workgroup; activations live in
// a per-wave [32][128] bf16 LDS tile; LN/pool/normalize in f32 on fragments.
// ---------------------------------------------------------------------------

typedef __attribute__((ext_vector_type(16))) __bf16 v16bf;
typedef __attribute__((ext_vector_type(8)))  float  v8f;

#define NCLUSTERS   32768
#define NWAVES      8
#define THREADS     256
#define ACT_STRIDE  128

// bf16-element offsets inside the LDS weight arena (all transposed [64][Kpad])
#define OFF_W1_0 0          // [64][32]  (K=10 zero-padded to 32)
#define OFF_W2_0 2048       // [64][64]
#define OFF_W1_1 6144       // [64][128]
#define OFF_W2_1 14336      // [64][64]
#define OFF_W1_2 18432      // [64][128]
#define OFF_W2_2 26624      // [64][64]
#define OFF_WL   30720      // [64][128]
#define WTS_ELEMS 38912

#define BIAS_ELEMS (13*64)  // b1,g,be,b2 per layer (x3) + bl, all f32

#define SMEM_BYTES (WTS_ELEMS*2 + BIAS_ELEMS*4 + NWAVES*32*ACT_STRIDE*2 + NWAVES*64*2)

struct Params {
  const float* x;
  float*       out;
  const float* w1[3];
  const float* w2[3];
  const float* wl;
  const float* bias[13]; // b1_0,g_0,be_0,b2_0, b1_1,..., b2_2, bl
};

union AccU  { v8f  v; float f[8]; };
union FragU { v16bf v; uint4 q[2]; };

__device__ __forceinline__ v16bf load_frag(const __bf16* p0, const __bf16* p1) {
  FragU u;
  u.q[0] = *(const uint4*)p0;
  u.q[1] = *(const uint4*)p1;
  return u.v;
}

__device__ __forceinline__ void init_acc(AccU acc[2][4], const float* bvec, int col) {
  #pragma unroll
  for (int nt = 0; nt < 4; ++nt) {
    const float bv = bvec[nt*16 + col];
    #pragma unroll
    for (int mt = 0; mt < 2; ++mt)
      #pragma unroll
      for (int j = 0; j < 8; ++j) acc[mt][nt].f[j] = bv;
  }
}

// D += A[32xK] * B[Kx64]; A from LDS act tile (stride 128), B from LDS
// transposed weights [64][kpad]. K = 32*KSTEPS.
template<int KSTEPS>
__device__ __forceinline__ void run_gemm(AccU acc[2][4], const __bf16* A,
                                         const __bf16* BT, int kpad, int lane)
{
  const int col = lane & 15;
  const int khA = ((lane >> 4) & 1) * 8;    // A: lanes 0-15 K 0-7/16-23, lanes 16-31 K 8-15/24-31
  const int khB = ((lane >> 4) & 1) * 8;    // B: lanes 0-15 K 0-7/16-23, lanes 16-31 K 8-15/24-31
  #pragma unroll
  for (int ks = 0; ks < KSTEPS; ++ks) {
    const int kbA = ks*32 + khA;
    const int kbB = ks*32 + khB;
    const __bf16* ar0 = A + col*ACT_STRIDE        + kbA;
    const __bf16* ar1 = A + (col + 16)*ACT_STRIDE + kbA;
    v16bf a0 = load_frag(ar0, ar0 + 16);
    v16bf a1 = load_frag(ar1, ar1 + 16);
    v16bf b[4];
    #pragma unroll
    for (int nt = 0; nt < 4; ++nt) {
      const __bf16* bp = BT + (nt*16 + col)*kpad + kbB;
      b[nt] = load_frag(bp, bp + 16);
    }
    #pragma unroll
    for (int nt = 0; nt < 4; ++nt) {
      acc[0][nt].v = __builtin_amdgcn_wmma_f32_16x16x32_bf16(
          false, a0, false, b[nt], (short)0, acc[0][nt].v, false, false);
      acc[1][nt].v = __builtin_amdgcn_wmma_f32_16x16x32_bf16(
          false, a1, false, b[nt], (short)0, acc[1][nt].v, false, false);
    }
  }
}

// LayerNorm(64) + ReLU on C fragments, store bf16 into act cols 0..63.
__device__ __forceinline__ void ln_relu_store(AccU acc[2][4], __bf16* A,
                                              const float* g, const float* be, int lane)
{
  const int col   = lane & 15;
  const int half8 = ((lane >> 4) & 1) * 8;
  float gv[4], bv[4];
  #pragma unroll
  for (int nt = 0; nt < 4; ++nt) { gv[nt] = g[nt*16 + col]; bv[nt] = be[nt*16 + col]; }
  #pragma unroll
  for (int mt = 0; mt < 2; ++mt)
    #pragma unroll
    for (int r = 0; r < 8; ++r) {
      float vals[4]; float s = 0.f, q = 0.f;
      #pragma unroll
      for (int nt = 0; nt < 4; ++nt) {
        const float t = acc[mt][nt].f[r]; vals[nt] = t; s += t; q += t*t;
      }
      // row reduction: xor masks < 16 stay within each 16-lane half (one row set)
      #pragma unroll
      for (int mk = 1; mk < 16; mk <<= 1) {
        s += __shfl_xor(s, mk, 32);
        q += __shfl_xor(q, mk, 32);
      }
      const float mu  = s * (1.0f/64.0f);
      const float var = q * (1.0f/64.0f) - mu*mu;
      const float rs  = rsqrtf(var + 1e-5f);
      const int   m   = mt*16 + r + half8;
      #pragma unroll
      for (int nt = 0; nt < 4; ++nt) {
        float t = (vals[nt] - mu) * rs * gv[nt] + bv[nt];
        t = fmaxf(t, 0.0f);
        A[m*ACT_STRIDE + nt*16 + col] = (__bf16)t;
      }
    }
}

// Store MLP output h (bf16) into act cols 0..63 and write per-column cluster
// max (segment_max over 32 rows) into pool[64].
__device__ __forceinline__ void store_h_pool(AccU acc[2][4], __bf16* A,
                                             __bf16* pool, int lane)
{
  const int col   = lane & 15;
  const int half8 = ((lane >> 4) & 1) * 8;
  float pm[4] = {-3.4e38f, -3.4e38f, -3.4e38f, -3.4e38f};
  #pragma unroll
  for (int mt = 0; mt < 2; ++mt)
    #pragma unroll
    for (int r = 0; r < 8; ++r) {
      const int m = mt*16 + r + half8;
      #pragma unroll
      for (int nt = 0; nt < 4; ++nt) {
        const float t = acc[mt][nt].f[r];
        A[m*ACT_STRIDE + nt*16 + col] = (__bf16)t;
        pm[nt] = fmaxf(pm[nt], t);
      }
    }
  #pragma unroll
  for (int nt = 0; nt < 4; ++nt)
    pm[nt] = fmaxf(pm[nt], __shfl_xor(pm[nt], 16, 32)); // combine both row halves
  if (lane < 16) {
    #pragma unroll
    for (int nt = 0; nt < 4; ++nt) pool[nt*16 + lane] = (__bf16)pm[nt];
  }
}

// Broadcast pool[64] into act cols 64..127 of every row (concat with h).
__device__ __forceinline__ void bcast_pool(__bf16* A, const __bf16* pool, int lane)
{
  const uint4* ps = (const uint4*)pool;                 // 64 bf16 = 8x16B
  uint4* pd = (uint4*)(A + lane*ACT_STRIDE + 64);       // lane == row
  #pragma unroll
  for (int j = 0; j < 8; ++j) pd[j] = ps[j];
}

// Final: cluster max-pool, L2 row-normalize, store f32 output row [64].
__device__ __forceinline__ void final_out(AccU acc[2][4], float* outp, int cl, int lane)
{
  const int col = lane & 15;
  float pm[4] = {-3.4e38f, -3.4e38f, -3.4e38f, -3.4e38f};
  #pragma unroll
  for (int mt = 0; mt < 2; ++mt)
    #pragma unroll
    for (int r = 0; r < 8; ++r)
      #pragma unroll
      for (int nt = 0; nt < 4; ++nt)
        pm[nt] = fmaxf(pm[nt], acc[mt][nt].f[r]);
  #pragma unroll
  for (int nt = 0; nt < 4; ++nt)
    pm[nt] = fmaxf(pm[nt], __shfl_xor(pm[nt], 16, 32));
  float ss = 0.f;
  #pragma unroll
  for (int nt = 0; nt < 4; ++nt) ss += pm[nt]*pm[nt];
  #pragma unroll
  for (int mk = 1; mk < 16; mk <<= 1) ss += __shfl_xor(ss, mk, 32);
  const float sc = 1.0f / fmaxf(sqrtf(ss), 1e-12f);
  if (lane < 16) {
    #pragma unroll
    for (int nt = 0; nt < 4; ++nt)
      outp[(size_t)cl*64 + nt*16 + col] = pm[nt]*sc;
  }
}

__global__ __launch_bounds__(THREADS)
void vectornet_kernel(Params P)
{
  extern __shared__ __align__(16) char smem[];
  __bf16* wts   = (__bf16*)smem;
  float*  biasS = (float*)(smem + WTS_ELEMS*2);
  __bf16* actB  = (__bf16*)(smem + WTS_ELEMS*2 + BIAS_ELEMS*4);
  __bf16* poolB = actB + NWAVES*32*ACT_STRIDE;

  const int tid = threadIdx.x;

  // ---- one-time per-WG: transpose weights fp32->bf16 into LDS (L2-resident reads)
  for (int e = tid; e < 64*32; e += THREADS) wts[OFF_W1_0 + e] = (__bf16)0.0f; // K-pad
  __syncthreads();
  auto tload = [&](const float* W, int K, int kpad, int off) {
    for (int e = tid; e < K*64; e += THREADS) {
      const int k = e >> 6, n = e & 63;
      wts[off + n*kpad + k] = (__bf16)W[e];
    }
  };
  tload(P.w1[0], 10,  32,  OFF_W1_0);
  tload(P.w2[0], 64,  64,  OFF_W2_0);
  tload(P.w1[1], 128, 128, OFF_W1_1);
  tload(P.w2[1], 64,  64,  OFF_W2_1);
  tload(P.w1[2], 128, 128, OFF_W1_2);
  tload(P.w2[2], 64,  64,  OFF_W2_2);
  tload(P.wl,    128, 128, OFF_WL);
  for (int j = 0; j < 13; ++j)
    for (int e = tid; e < 64; e += THREADS) biasS[j*64 + e] = P.bias[j][e];
  __syncthreads();

  // ---- per-wave cluster processing
  const int wave = tid >> 5;
  const int lane = tid & 31;
  const int cl   = blockIdx.x * NWAVES + wave;
  __bf16* A    = actB  + wave * 32 * ACT_STRIDE;
  __bf16* pool = poolB + wave * 64;
  const int col = lane & 15;

  // load x[cl*32 + lane][0..9] -> act row, K-padded to 32
  {
    const float* xr = P.x + (size_t)(cl*32 + lane) * 10;
    __bf16* ar = A + lane*ACT_STRIDE;
    #pragma unroll
    for (int k = 0; k < 10; ++k) ar[k] = (__bf16)xr[k];
    #pragma unroll
    for (int k = 10; k < 32; ++k) ar[k] = (__bf16)0.0f;
  }
  __syncthreads();

  AccU acc[2][4];

  // ---------------- layer 0 ----------------
  init_acc(acc, biasS + 0, col);                         // b1_0
  run_gemm<1>(acc, A, wts + OFF_W1_0, 32, lane);
  ln_relu_store(acc, A, biasS + 64, biasS + 128, lane);  // g_0, be_0
  __syncthreads();
  init_acc(acc, biasS + 192, col);                       // b2_0
  run_gemm<2>(acc, A, wts + OFF_W2_0, 64, lane);
  store_h_pool(acc, A, pool, lane);
  __syncthreads();
  bcast_pool(A, pool, lane);
  __syncthreads();

  // ---------------- layer 1 ----------------
  init_acc(acc, biasS + 256, col);                       // b1_1
  run_gemm<4>(acc, A, wts + OFF_W1_1, 128, lane);
  ln_relu_store(acc, A, biasS + 256 + 64, biasS + 256 + 128, lane);
  __syncthreads();
  init_acc(acc, biasS + 256 + 192, col);                 // b2_1
  run_gemm<2>(acc, A, wts + OFF_W2_1, 64, lane);
  store_h_pool(acc, A, pool, lane);
  __syncthreads();
  bcast_pool(A, pool, lane);
  __syncthreads();

  // ---------------- layer 2 ----------------
  init_acc(acc, biasS + 512, col);                       // b1_2
  run_gemm<4>(acc, A, wts + OFF_W1_2, 128, lane);
  ln_relu_store(acc, A, biasS + 512 + 64, biasS + 512 + 128, lane);
  __syncthreads();
  init_acc(acc, biasS + 512 + 192, col);                 // b2_2
  run_gemm<2>(acc, A, wts + OFF_W2_2, 64, lane);
  store_h_pool(acc, A, pool, lane);
  __syncthreads();
  bcast_pool(A, pool, lane);
  __syncthreads();

  // ---------------- final linear + pool + normalize ----------------
  init_acc(acc, biasS + 768, col);                       // bl
  run_gemm<4>(acc, A, wts + OFF_WL, 128, lane);
  final_out(acc, P.out, cl, lane);
}

extern "C" void kernel_launch(void* const* d_in, const int* in_sizes, int n_in,
                              void* d_out, int out_size, void* d_ws, size_t ws_size,
                              hipStream_t stream) {
  (void)in_sizes; (void)n_in; (void)d_ws; (void)ws_size; (void)out_size;

  Params P;
  P.x   = (const float*)d_in[0];
  P.out = (float*)d_out;
  // dict order: x, cluster, n_clusters, then per layer {w1,b1,g,be,w2,b2}, wl, bl
  P.w1[0] = (const float*)d_in[3];   P.w2[0] = (const float*)d_in[7];
  P.w1[1] = (const float*)d_in[9];   P.w2[1] = (const float*)d_in[13];
  P.w1[2] = (const float*)d_in[15];  P.w2[2] = (const float*)d_in[19];
  P.wl    = (const float*)d_in[21];
  const int bidx[13] = {4,5,6,8, 10,11,12,14, 16,17,18,20, 22};
  for (int j = 0; j < 13; ++j) P.bias[j] = (const float*)d_in[bidx[j]];

  // >64KB dynamic LDS: opt in every call (idempotent, deterministic)
  hipFuncSetAttribute(reinterpret_cast<const void*>(vectornet_kernel),
                      hipFuncAttributeMaxDynamicSharedMemorySize, SMEM_BYTES);

  vectornet_kernel<<<NCLUSTERS / NWAVES, THREADS, SMEM_BYTES, stream>>>(P);
}
